// E_GCL_59871844106306
// MI455X (gfx1250) — compile-verified, matrix operand order checked by reading, not run on hardware
//
#include <hip/hip_runtime.h>
#include <hip/hip_fp16.h>

typedef __attribute__((ext_vector_type(16))) _Float16 v16h;
typedef __attribute__((ext_vector_type(8)))  _Float16 half8;
typedef __attribute__((ext_vector_type(8)))  float    v8f;

#define DIM   128
#define WAVES 8

__device__ __forceinline__ float siluf(float x){ return x * (1.0f/(1.0f+__expf(-x))); }
__device__ __forceinline__ float sigmf(float x){ return 1.0f/(1.0f+__expf(-x)); }

__device__ __forceinline__ v8f vzero(){
  v8f z;
#pragma unroll
  for (int i=0;i<8;i++) z[i]=0.0f;
  return z;
}

__device__ __forceinline__ v8f wmma_f16(v16h a, v16h b, v8f c){
  // D = A(16x32 f16) * B(32x16 f16) + C(16x16 f32)
  return __builtin_amdgcn_wmma_f32_16x16x32_f16(false, a, false, b, (short)0, c, false, false);
}

// Build A/B fragment (16x32 f16 layout, wave32):
// lane L: elements 0..7 = K klo+0..7, elements 8..15 = K klo+16..23, klo = (L<16?0:8)
__device__ __forceinline__ v16h frag_from_f32(const float* __restrict__ src){
  float4 f0 = *(const float4*)(src + 0);
  float4 f1 = *(const float4*)(src + 4);
  float4 f2 = *(const float4*)(src + 16);
  float4 f3 = *(const float4*)(src + 20);
  v16h a;
  a[0]=(_Float16)f0.x;  a[1]=(_Float16)f0.y;  a[2]=(_Float16)f0.z;  a[3]=(_Float16)f0.w;
  a[4]=(_Float16)f1.x;  a[5]=(_Float16)f1.y;  a[6]=(_Float16)f1.z;  a[7]=(_Float16)f1.w;
  a[8]=(_Float16)f2.x;  a[9]=(_Float16)f2.y;  a[10]=(_Float16)f2.z; a[11]=(_Float16)f2.w;
  a[12]=(_Float16)f3.x; a[13]=(_Float16)f3.y; a[14]=(_Float16)f3.z; a[15]=(_Float16)f3.w;
  return a;
}

__device__ __forceinline__ v16h frag_from_f16(const _Float16* p){
  half8 lo = *(const half8*)p;
  half8 hi = *(const half8*)(p + 16);
  v16h a;
#pragma unroll
  for (int i=0;i<8;i++){ a[i]=lo[i]; a[8+i]=hi[i]; }
  return a;
}

// ---------------------------------------------------------------- zero scratch
__global__ void egcl_zero(float* __restrict__ p, long n){
  long i = (long)blockIdx.x*blockDim.x + threadIdx.x;
  long stride = (long)gridDim.x*blockDim.x;
  for (; i < n; i += stride) p[i] = 0.0f;
}

// ------------------------------------------------- weight transpose + f16 cvt
__global__ void egcl_prep(const float* __restrict__ We1, const float* __restrict__ We2,
                          const float* __restrict__ Wc1, const float* __restrict__ Wn1,
                          const float* __restrict__ Wn2,
                          _Float16* __restrict__ Wt1, _Float16* __restrict__ Wt2,
                          _Float16* __restrict__ Wtc1, _Float16* __restrict__ Wtn1,
                          _Float16* __restrict__ Wtn2, float* __restrict__ w_r){
  int idx = blockIdx.x*blockDim.x + threadIdx.x;
  if (idx < 32768){            // We1 (257x128) rows 0..255 -> Wt1[128][256]
    int o = idx >> 8, i = idx & 255;
    Wt1[o*256 + i] = (_Float16)We1[i*128 + o];
  } else if (idx < 49152){     // We2 (128x128) -> Wt2[128][128]
    int l = idx - 32768; int o = l >> 7, i = l & 127;
    Wt2[o*128 + i] = (_Float16)We2[i*128 + o];
  } else if (idx < 65536){     // Wc1
    int l = idx - 49152; int o = l >> 7, i = l & 127;
    Wtc1[o*128 + i] = (_Float16)Wc1[i*128 + o];
  } else if (idx < 98304){     // Wn1 (256x128) -> Wtn1[128][256]
    int l = idx - 65536; int o = l >> 8, i = l & 255;
    Wtn1[o*256 + i] = (_Float16)Wn1[i*128 + o];
  } else if (idx < 114688){    // Wn2
    int l = idx - 98304; int o = l >> 7, i = l & 127;
    Wtn2[o*128 + i] = (_Float16)Wn2[i*128 + o];
  } else if (idx < 114816){    // radial row of We1 kept in f32
    int j = idx - 114688;
    w_r[j] = We1[256*128 + j];
  }
}

// ------------------------------------------------------------------ edge model
__global__ __launch_bounds__(256) void egcl_edge(
    const float* __restrict__ h, const int* __restrict__ erow, const int* __restrict__ ecol,
    const float* __restrict__ coord,
    const _Float16* __restrict__ gWt1, const _Float16* __restrict__ gWt2,
    const _Float16* __restrict__ gWtc,
    const float* __restrict__ w_r, const float* __restrict__ be1, const float* __restrict__ be2,
    const float* __restrict__ Watt, const float* __restrict__ batt,
    const float* __restrict__ bc1, const float* __restrict__ Wc2,
    float* __restrict__ out_e, float* __restrict__ agg,
    float* __restrict__ coordsum, float* __restrict__ cnt, int E)
{
  extern __shared__ char smem[];
  _Float16* sWt1 = (_Float16*)smem;            // 128*256 f16
  _Float16* sWt2 = sWt1 + DIM*256;             // 128*128
  _Float16* sWtc = sWt2 + DIM*DIM;             // 128*128
  _Float16* sStg = sWtc + DIM*DIM;             // WAVES*16*128
  float*    sCw  = (float*)(sStg + WAVES*16*DIM); // WAVES*16

  { // cooperative weight staging into LDS (16B vectors)
    uint4* d = (uint4*)sWt1; const uint4* s = (const uint4*)gWt1;
    for (int i=threadIdx.x; i<(DIM*256)/8; i+=blockDim.x) d[i]=s[i];
    d = (uint4*)sWt2; s = (const uint4*)gWt2;
    for (int i=threadIdx.x; i<(DIM*DIM)/8; i+=blockDim.x) d[i]=s[i];
    d = (uint4*)sWtc; s = (const uint4*)gWtc;
    for (int i=threadIdx.x; i<(DIM*DIM)/8; i+=blockDim.x) d[i]=s[i];
  }
  __syncthreads();

  const int lane = threadIdx.x & 31;
  const int wave = threadIdx.x >> 5;
  const int m    = lane & 15;       // edge (A-row) / output column within tile
  const int hlf  = lane >> 4;
  const int klo  = hlf * 8;
  const long tile = (long)blockIdx.x * WAVES + wave;
  const long e    = tile * 16 + m;
  const long eL   = (e < E) ? e : (long)(E - 1);
  const int row = erow[eL];
  const int col = ecol[eL];

  const float cdx = coord[row*3+0] - coord[col*3+0];
  const float cdy = coord[row*3+1] - coord[col*3+1];
  const float cdz = coord[row*3+2] - coord[col*3+2];
  const float radial = cdx*cdx + cdy*cdy + cdz*cdz;

  _Float16* myStg = sStg + wave * 16 * DIM;

  // ---- GEMM1: [h[row] | h[col]] (16x256) @ We1[0:256,:] ----
  v8f acc[8];
#pragma unroll
  for (int t=0;t<8;t++) acc[t] = vzero();
#pragma unroll
  for (int kk=0;kk<8;kk++){
    const float* src = (kk < 4) ? (h + (size_t)row*DIM + kk*32 + klo)
                                : (h + (size_t)col*DIM + (kk-4)*32 + klo);
    v16h a = frag_from_f32(src);
#pragma unroll
    for (int t=0;t<8;t++){
      v16h b = frag_from_f16(sWt1 + (size_t)(t*16+m)*256 + kk*32 + klo);
      acc[t] = wmma_f16(a, b, acc[t]);
    }
  }

  float radr[8];
#pragma unroll
  for (int r=0;r<8;r++) radr[r] = __shfl(radial, r + hlf*8, 32);

  // rank-1 radial update + bias + SiLU, stage f16 tile for GEMM2
#pragma unroll
  for (int t=0;t<8;t++){
    const int n = t*16 + m;
    const float wr = w_r[n], b1 = be1[n];
#pragma unroll
    for (int r=0;r<8;r++){
      float x = siluf(acc[t][r] + radr[r]*wr + b1);
      acc[t][r] = x;
      myStg[(size_t)(r + hlf*8)*DIM + n] = (_Float16)x;
    }
  }
  __syncthreads();

  // ---- GEMM2: e (16x128) @ We2 ----
  v8f f[8];
#pragma unroll
  for (int t=0;t<8;t++) f[t] = vzero();
#pragma unroll
  for (int kk=0;kk<4;kk++){
    v16h a = frag_from_f16(myStg + (size_t)m*DIM + kk*32 + klo);
#pragma unroll
    for (int t=0;t<8;t++){
      v16h b = frag_from_f16(sWt2 + (size_t)(t*16+m)*DIM + kk*32 + klo);
      f[t] = wmma_f16(a, b, f[t]);
    }
  }
  float wattn[8];
#pragma unroll
  for (int t=0;t<8;t++){
    const int n = t*16 + m;
    const float b2 = be2[n];
    wattn[t] = Watt[n];
#pragma unroll
    for (int r=0;r<8;r++) f[t][r] = siluf(f[t][r] + b2);
  }

  // attention: sigmoid(f . Watt + batt), half-wave reduction matches C layout
  const float b0 = batt[0];
#pragma unroll
  for (int r=0;r<8;r++){
    float s = 0.0f;
#pragma unroll
    for (int t=0;t<8;t++) s += f[t][r] * wattn[t];
    s += __shfl_xor(s, 1, 16); s += __shfl_xor(s, 2, 16);
    s += __shfl_xor(s, 4, 16); s += __shfl_xor(s, 8, 16);
    const float at = sigmf(s + b0);
#pragma unroll
    for (int t=0;t<8;t++) f[t][r] *= at;
  }

  // stage f16 edge_feat for GEMM3; write edge_feat output; scatter agg
  int rowr[8];
#pragma unroll
  for (int r=0;r<8;r++) rowr[r] = __shfl(row, r + hlf*8, 32);

#pragma unroll
  for (int t=0;t<8;t++){
    const int n = t*16 + m;
#pragma unroll
    for (int r=0;r<8;r++){
      const float v = f[t][r];
      myStg[(size_t)(r + hlf*8)*DIM + n] = (_Float16)v;
      const long ee = tile*16 + r + hlf*8;
      if (ee < E){
        out_e[ee*(long)DIM + n] = v;
        unsafeAtomicAdd(&agg[(size_t)rowr[r]*DIM + n], v);
      }
    }
  }
  __syncthreads();

  // ---- GEMM3: f (16x128) @ Wc1, SiLU, dot Wc2 -> coord weight ----
  v8f g[8];
#pragma unroll
  for (int t=0;t<8;t++) g[t] = vzero();
#pragma unroll
  for (int kk=0;kk<4;kk++){
    v16h a = frag_from_f16(myStg + (size_t)m*DIM + kk*32 + klo);
#pragma unroll
    for (int t=0;t<8;t++){
      v16h b = frag_from_f16(sWtc + (size_t)(t*16+m)*DIM + kk*32 + klo);
      g[t] = wmma_f16(a, b, g[t]);
    }
  }
  float bc1n[8], wc2n[8];
#pragma unroll
  for (int t=0;t<8;t++){ const int n=t*16+m; bc1n[t]=bc1[n]; wc2n[t]=Wc2[n]; }

  float cwv[8];
#pragma unroll
  for (int r=0;r<8;r++){
    float s = 0.0f;
#pragma unroll
    for (int t=0;t<8;t++) s += siluf(g[t][r] + bc1n[t]) * wc2n[t];
    s += __shfl_xor(s, 1, 16); s += __shfl_xor(s, 2, 16);
    s += __shfl_xor(s, 4, 16); s += __shfl_xor(s, 8, 16);
    cwv[r] = s;
  }
  // distribute cw[m'] to its edge lane via LDS
#pragma unroll
  for (int r=0;r<8;r++)
    if (m == r) sCw[wave*16 + r + hlf*8] = cwv[r];
  __syncthreads();
  const float cw = sCw[wave*16 + m];

  if (hlf == 0 && e < E){
    float tx = fminf(fmaxf(cdx*cw, -10.0f), 10.0f);
    float ty = fminf(fmaxf(cdy*cw, -10.0f), 10.0f);
    float tz = fminf(fmaxf(cdz*cw, -10.0f), 10.0f);
    unsafeAtomicAdd(&coordsum[(size_t)row*3+0], tx);
    unsafeAtomicAdd(&coordsum[(size_t)row*3+1], ty);
    unsafeAtomicAdd(&coordsum[(size_t)row*3+2], tz);
    unsafeAtomicAdd(&cnt[row], 1.0f);
  }
}

// ------------------------------------------------------------------ node model
__global__ __launch_bounds__(256) void egcl_node(
    const float* __restrict__ h, const float* __restrict__ agg,
    const _Float16* __restrict__ gW1, const _Float16* __restrict__ gW2,
    const float* __restrict__ bn1, const float* __restrict__ bn2,
    float* __restrict__ out_h, int N)
{
  extern __shared__ char smem[];
  _Float16* sW1 = (_Float16*)smem;          // 128*256
  _Float16* sW2 = sW1 + DIM*256;            // 128*128
  _Float16* sStg = sW2 + DIM*DIM;           // WAVES*16*128

  {
    uint4* d = (uint4*)sW1; const uint4* s = (const uint4*)gW1;
    for (int i=threadIdx.x; i<(DIM*256)/8; i+=blockDim.x) d[i]=s[i];
    d = (uint4*)sW2; s = (const uint4*)gW2;
    for (int i=threadIdx.x; i<(DIM*DIM)/8; i+=blockDim.x) d[i]=s[i];
  }
  __syncthreads();

  const int lane = threadIdx.x & 31;
  const int wave = threadIdx.x >> 5;
  const int m    = lane & 15;
  const int hlf  = lane >> 4;
  const int klo  = hlf * 8;
  const long tile = (long)blockIdx.x * WAVES + wave;
  const long node = tile * 16 + m;
  const long nodeL = (node < N) ? node : (long)(N - 1);

  _Float16* myStg = sStg + wave * 16 * DIM;

  // ---- GEMM1: [h | agg] (16x256) @ Wn1 ----
  v8f acc[8];
#pragma unroll
  for (int t=0;t<8;t++) acc[t] = vzero();
#pragma unroll
  for (int kk=0;kk<8;kk++){
    const float* src = (kk < 4) ? (h   + (size_t)nodeL*DIM + kk*32 + klo)
                                : (agg + (size_t)nodeL*DIM + (kk-4)*32 + klo);
    v16h a = frag_from_f32(src);
#pragma unroll
    for (int t=0;t<8;t++){
      v16h b = frag_from_f16(sW1 + (size_t)(t*16+m)*256 + kk*32 + klo);
      acc[t] = wmma_f16(a, b, acc[t]);
    }
  }
#pragma unroll
  for (int t=0;t<8;t++){
    const int n = t*16 + m;
    const float b1 = bn1[n];
#pragma unroll
    for (int r=0;r<8;r++){
      float x = siluf(acc[t][r] + b1);
      myStg[(size_t)(r + hlf*8)*DIM + n] = (_Float16)x;
    }
  }
  __syncthreads();

  // ---- GEMM2 + residual ----
  v8f f[8];
#pragma unroll
  for (int t=0;t<8;t++) f[t] = vzero();
#pragma unroll
  for (int kk=0;kk<4;kk++){
    v16h a = frag_from_f16(myStg + (size_t)m*DIM + kk*32 + klo);
#pragma unroll
    for (int t=0;t<8;t++){
      v16h b = frag_from_f16(sW2 + (size_t)(t*16+m)*DIM + kk*32 + klo);
      f[t] = wmma_f16(a, b, f[t]);
    }
  }
#pragma unroll
  for (int t=0;t<8;t++){
    const int n = t*16 + m;
    const float b2 = bn2[n];
#pragma unroll
    for (int r=0;r<8;r++){
      const long nr = tile*16 + r + hlf*8;
      if (nr < N)
        out_h[nr*(long)DIM + n] = f[t][r] + b2 + h[(size_t)nr*DIM + n];
    }
  }
}

// ---------------------------------------------------------------- coord output
__global__ void egcl_coord(const float* __restrict__ coord, const float* __restrict__ coordsum,
                           const float* __restrict__ cnt, float* __restrict__ out_c, int N){
  int i = blockIdx.x*blockDim.x + threadIdx.x;
  if (i < N){
    const float c = fmaxf(cnt[i], 1.0f);
#pragma unroll
    for (int k=0;k<3;k++){
      float mv = coordsum[(size_t)i*3 + k] / c;
      mv = fminf(fmaxf(mv, -10.0f), 10.0f);
      out_c[(size_t)i*3 + k] = coord[(size_t)i*3 + k] + mv;
    }
  }
}

// ------------------------------------------------------------------- launcher
extern "C" void kernel_launch(void* const* d_in, const int* in_sizes, int n_in,
                              void* d_out, int out_size, void* d_ws, size_t ws_size,
                              hipStream_t stream) {
  const float* h     = (const float*)d_in[0];
  const int*   ei    = (const int*)d_in[1];     // (2,E): rows then cols
  const float* coord = (const float*)d_in[2];
  const float* We1   = (const float*)d_in[3];
  const float* be1   = (const float*)d_in[4];
  const float* We2   = (const float*)d_in[5];
  const float* be2   = (const float*)d_in[6];
  const float* Watt  = (const float*)d_in[7];
  const float* batt  = (const float*)d_in[8];
  const float* Wc1   = (const float*)d_in[9];
  const float* bc1   = (const float*)d_in[10];
  const float* Wc2   = (const float*)d_in[11];
  const float* Wn1   = (const float*)d_in[12];
  const float* bn1   = (const float*)d_in[13];
  const float* Wn2   = (const float*)d_in[14];
  const float* bn2   = (const float*)d_in[15];

  const int N = in_sizes[0] / DIM;
  const int E = in_sizes[1] / 2;

  // workspace carve-up
  float* agg      = (float*)d_ws;                 // N*128
  float* coordsum = agg + (size_t)N*DIM;          // N*3
  float* cnt      = coordsum + (size_t)N*3;       // N
  _Float16* Wt1h  = (_Float16*)(cnt + N);         // 128*256
  _Float16* Wt2h  = Wt1h + DIM*256;               // 128*128
  _Float16* Wtc1h = Wt2h + DIM*DIM;               // 128*128
  _Float16* Wtn1h = Wtc1h + DIM*DIM;              // 128*256
  _Float16* Wtn2h = Wtn1h + DIM*256;              // 128*128
  float* w_r      = (float*)(Wtn2h + DIM*DIM);    // 128

  float* out_h = (float*)d_out;
  float* out_c = out_h + (size_t)N*DIM;
  float* out_e = out_c + (size_t)N*3;

  // 1) zero accumulators (agg + coordsum + cnt = N*132 floats)
  egcl_zero<<<2048, 256, 0, stream>>>(agg, (long)N*132);

  // 2) transpose + f16-convert weights
  egcl_prep<<<(114816 + 255)/256, 256, 0, stream>>>(
      We1, We2, Wc1, Wn1, Wn2, Wt1h, Wt2h, Wtc1h, Wtn1h, Wtn2h, w_r);

  // 3) fused edge model (WMMA) + scatter
  const int etiles  = (E + 15) / 16;
  const int eblocks = (etiles + WAVES - 1) / WAVES;
  const size_t elds = (size_t)(DIM*256 + DIM*DIM + DIM*DIM + WAVES*16*DIM)*2 + WAVES*16*4;
  egcl_edge<<<eblocks, 256, elds, stream>>>(
      h, ei, ei + E, coord, Wt1h, Wt2h, Wtc1h, w_r, be1, be2, Watt, batt,
      bc1, Wc2, out_e, agg, coordsum, cnt, E);

  // 4) node model (WMMA) + residual
  const int ntiles  = (N + 15) / 16;
  const int nblocks = (ntiles + WAVES - 1) / WAVES;
  const size_t nlds = (size_t)(DIM*256 + DIM*DIM + WAVES*16*DIM)*2;
  egcl_node<<<nblocks, 256, nlds, stream>>>(h, agg, Wtn1h, Wtn2h, bn1, bn2, out_h, N);

  // 5) coord segment-mean + clip
  egcl_coord<<<(N + 255)/256, 256, 0, stream>>>(coord, coordsum, cnt, out_c, N);
}
